// GCQEncoder_12567074308774
// MI455X (gfx1250) — matrix operand order repaired and checked by reading, not compiled, not run on hardware
//
#include <hip/hip_runtime.h>

typedef __bf16 bf16;
typedef __bf16 v16bf __attribute__((ext_vector_type(16)));
typedef float  v8f   __attribute__((ext_vector_type(8)));

#define B_       4
#define NG       1024
#define NC       1024
#define NQ       256
#define FDIM     512
#define NHEADS_  8
#define HDIM_    64
#define HD       512     // NHEADS_*HDIM_ == FDIM (convenient)
#define FF       (FDIM*FDIM)
#define ALPHA_   0.2f
#define NEGBIG   -9e15f

// ---------------------------------------------------------------------------
// Batched NT GEMM, bf16 inputs, f32 out:
//   C[z][m][n] (+)= sum_k A[z][m][k] * Bt[z][n][k]
// z = zo*inner + zi with independent (outer, inner) batch strides.
// Block: 256 threads = 8 wave32 waves; tile 128(M) x TNN(N), K-step 32.
// TNN=128: each wave computes 32x64 = 2x4 WMMA tiles (8 wmma / K-step).
// TNN=64 : each wave computes 32x32 = 2x2 WMMA tiles (att@Wk, N=64).
// ---------------------------------------------------------------------------
#define TM 128
#define TKK 32
#define PAD_ROW 40   // bf16 elements per LDS row (32 data + 8 pad, keeps 16B align)

template <int TNN>
__global__ __launch_bounds__(256)
void k_gemm_bf16_nt(const bf16* __restrict__ A, const bf16* __restrict__ Bt,
                    float* __restrict__ C, int K,
                    long aO, long aI, int lda,
                    long bO, long bI, int ldb,
                    long cO, long cI, int ldc,
                    int inner, int accumulate)
{
    constexpr int NI = TNN / 32;   // WMMA tiles per wave along N (2 or 4)
    __shared__ bf16 sA[TM  * PAD_ROW];
    __shared__ bf16 sB[TNN * PAD_ROW];

    const int z  = blockIdx.z;
    const int zo = z / inner, zi = z - zo * inner;
    A  += zo * aO + zi * aI;
    Bt += zo * bO + zi * bI;
    C  += zo * cO + zi * cI;

    const int m0   = blockIdx.x * TM;
    const int n0   = blockIdx.y * TNN;
    const int tid  = threadIdx.x;
    const int lane = tid & 31;
    const int wave = tid >> 5;
    const int wm   = (wave & 3) * 32;          // 4 waves along M
    const int wn   = (wave >> 2) * (TNN / 2);  // 2 waves along N
    const int l16  = lane & 15;
    const int half = lane >> 4;                // 0: lanes 0-15, 1: lanes 16-31

    // global->LDS staging assignment
    const int ar = tid >> 1, ac = (tid & 1) * 16;  // A: 128 rows x 32, 16 bf16/thread
    int br, bc;
    if constexpr (TNN == 128) { br = tid >> 1; bc = (tid & 1) * 16; }
    else                      { br = tid >> 2; bc = (tid & 3) * 8;  }

    const bf16* gA = A  + (long)(m0 + ar) * lda + ac;
    const bf16* gB = Bt + (long)(n0 + br) * ldb + bc;

    v8f acc[2][NI] = {};

    for (int k0 = 0; k0 < K; k0 += TKK) {
        uint4 a0 = ((const uint4*)(gA + k0))[0];
        uint4 a1 = ((const uint4*)(gA + k0))[1];
        uint4 b0 = ((const uint4*)(gB + k0))[0];
        uint4 b1;
        if constexpr (TNN == 128) b1 = ((const uint4*)(gB + k0))[1];
        if (k0 + TKK < K) {  // lowers to global_prefetch_b8 on gfx1250
            __builtin_prefetch((const void*)(gA + k0 + TKK), 0, 1);
            __builtin_prefetch((const void*)(gB + k0 + TKK), 0, 1);
        }
        __syncthreads();
        *(uint4*)&sA[ar * PAD_ROW + ac]     = a0;
        *(uint4*)&sA[ar * PAD_ROW + ac + 8] = a1;
        *(uint4*)&sB[br * PAD_ROW + bc]     = b0;
        if constexpr (TNN == 128) *(uint4*)&sB[br * PAD_ROW + bc + 8] = b1;
        __syncthreads();

        // A 16x32 bf16 fragment: lanes 0-15 row M=l16 hold K{0..7,16..23};
        // lanes 16-31 hold K{8..15,24..31}. Two aligned b128 LDS reads each.
        v16bf aF[2], bF[NI];
#pragma unroll
        for (int i = 0; i < 2; ++i) {
            int row = wm + i * 16 + l16;
            union { uint4 u[2]; v16bf v; } t;
            t.u[0] = *(const uint4*)&sA[row * PAD_ROW + half * 8];
            t.u[1] = *(const uint4*)&sA[row * PAD_ROW + half * 8 + 16];
            aF[i] = t.v;
        }
        // B 32x16 fragment: lane holds column N=l16; lanes 0-15 K{0..7,16..23},
        // lanes 16-31 K{8..15,24..31} (same K order as A).
#pragma unroll
        for (int j = 0; j < NI; ++j) {
            int col = wn + j * 16 + l16;
            union { uint4 u[2]; v16bf v; } t;
            t.u[0] = *(const uint4*)&sB[col * PAD_ROW + half * 8];
            t.u[1] = *(const uint4*)&sB[col * PAD_ROW + half * 8 + 16];
            bF[j] = t.v;
        }
#pragma unroll
        for (int i = 0; i < 2; ++i)
#pragma unroll
            for (int j = 0; j < NI; ++j)
                acc[i][j] = __builtin_amdgcn_wmma_f32_16x16x32_bf16(
                    false, aF[i], false, bF[j], (short)0, acc[i][j], false, false);
    }

    // C layout: VGPR r -> M = r (lanes 0-15) / r+8 (lanes 16-31); N = l16.
#pragma unroll
    for (int i = 0; i < 2; ++i) {
#pragma unroll
        for (int j = 0; j < NI; ++j) {
            int rbase = m0 + wm + i * 16 + half * 8;
            int col   = n0 + wn + j * 16 + l16;
#pragma unroll
            for (int r = 0; r < 8; ++r) {
                long idx = (long)(rbase + r) * ldc + col;
                if (accumulate) C[idx] += acc[i][j][r];
                else            C[idx]  = acc[i][j][r];
            }
        }
    }
}

// ---------------------------------------------------------------------------
// Elementwise / layout helpers
// ---------------------------------------------------------------------------
__global__ void k_f32_to_bf16(const float* __restrict__ in, bf16* __restrict__ out, long n) {
    long i = (long)blockIdx.x * blockDim.x + threadIdx.x;
    if (i < n) out[i] = (bf16)in[i];
}

// out[b][c][r] = (bf16) in[b][r][c]   (in is [nb, R, C] row-major)
template <typename T>
__global__ void k_transpose_convert(const T* __restrict__ in, bf16* __restrict__ out,
                                    int R, int C, long total) {
    long i = (long)blockIdx.x * blockDim.x + threadIdx.x;
    if (i >= total) return;
    int  c = (int)(i % C);
    long t = i / C;
    int  r = (int)(t % R);
    long b = t / R;
    out[b * (long)R * C + (long)c * R + r] = (bf16)(float)in[i];
}

// W [H, F, D] f32  ->  out [H*D, F] bf16 (transposed B operand for projections)
__global__ void k_pack_attw(const float* __restrict__ W, bf16* __restrict__ out) {
    int i = blockIdx.x * blockDim.x + threadIdx.x;
    if (i >= NHEADS_ * FDIM * HDIM_) return;
    int d = i % HDIM_;
    int f = (i / HDIM_) % FDIM;
    int h = i / (HDIM_ * FDIM);
    out[(long)(h * HDIM_ + d) * FDIM + f] = (bf16)W[i];
}

// s[(b*H+h)*N + n] = sum_d P[b, n, h*64+d] * a[h, d]
__global__ void k_dotvec(const float* __restrict__ P, const float* __restrict__ a,
                         float* __restrict__ s, int N, long total) {
    long i = (long)blockIdx.x * blockDim.x + threadIdx.x;
    if (i >= total) return;
    int  n = (int)(i % N);
    int  h = (int)((i / N) % NHEADS_);
    long b = i / ((long)N * NHEADS_);
    const float* p  = P + (b * N + n) * (long)HD + h * HDIM_;
    const float* av = a + h * HDIM_;
    float acc = 0.f;
#pragma unroll 8
    for (int d = 0; d < HDIM_; ++d) acc += p[d] * av[d];
    s[i] = acc;
}

__device__ __forceinline__ float lrelu(float x) { return x > 0.f ? x : ALPHA_ * x; }

// per (b,h,m): max_n and sum_n exp of masked leaky-relu scores (softmax over query axis)
__global__ void k_colstats(const float* __restrict__ s1, const float* __restrict__ s2,
                           const int* __restrict__ adj, long adjB, int sN, int sM,
                           float* __restrict__ cmax, float* __restrict__ csum,
                           int N, int M, long total) {
    long i = (long)blockIdx.x * blockDim.x + threadIdx.x;
    if (i >= total) return;
    int  m = (int)(i % M);
    int  h = (int)((i / M) % NHEADS_);
    long b = i / ((long)M * NHEADS_);
    const float* s1p = s1 + (b * NHEADS_ + h) * (long)N;
    float s2m = s2[i];
    const int* ap = adj + b * adjB + (long)m * sM;
    float mx = -3.4e38f;
    for (int n = 0; n < N; ++n) {
        float e = ap[(long)n * sN] > 0 ? lrelu(s1p[n] + s2m) : NEGBIG;
        mx = fmaxf(mx, e);
    }
    float sum = 0.f;
    for (int n = 0; n < N; ++n) {
        float e = ap[(long)n * sN] > 0 ? lrelu(s1p[n] + s2m) : NEGBIG;
        sum += __expf(e - mx);
    }
    cmax[i] = mx;
    csum[i] = sum;
}

// att[(b*H+h)*N*M + n*M + m] in bf16, normalized over n
__global__ void k_att(const float* __restrict__ s1, const float* __restrict__ s2,
                      const int* __restrict__ adj, long adjB, int sN, int sM,
                      const float* __restrict__ cmax, const float* __restrict__ csum,
                      bf16* __restrict__ att, int N, int M, long total) {
    long i = (long)blockIdx.x * blockDim.x + threadIdx.x;
    if (i >= total) return;
    int  m = (int)(i % M);
    long t = i / M;
    int  n = (int)(t % N);
    int  h = (int)((t / N) % NHEADS_);
    long b = t / ((long)N * NHEADS_);
    long bh = b * NHEADS_ + h;
    int a = adj[b * adjB + (long)n * sN + (long)m * sM];
    float e = a > 0 ? lrelu(s1[bh * N + n] + s2[bh * M + m]) : NEGBIG;
    float v = __expf(e - cmax[bh * M + m]) / csum[bh * M + m];
    att[i] = (bf16)v;
}

__global__ void k_elu(float* __restrict__ x, long n) {
    long i = (long)blockIdx.x * blockDim.x + threadIdx.x;
    if (i < n) { float v = x[i]; x[i] = v > 0.f ? v : (__expf(v) - 1.f); }
}

// out = sigmoid(gate) * newa + (1 - sigmoid(gate)) * a
__global__ void k_fusion_ep(const float* __restrict__ a, const float* __restrict__ newa,
                            const float* __restrict__ gate, float* __restrict__ out, long n) {
    long i = (long)blockIdx.x * blockDim.x + threadIdx.x;
    if (i < n) {
        float f = 1.f / (1.f + __expf(-gate[i]));
        out[i] = f * newa[i] + (1.f - f) * a[i];
    }
}

// ---------------------------------------------------------------------------
// Host orchestration
// ---------------------------------------------------------------------------
extern "C" void kernel_launch(void* const* d_in, const int* in_sizes, int n_in,
                              void* d_out, int out_size, void* d_ws, size_t ws_size,
                              hipStream_t stream) {
    (void)in_sizes; (void)n_in; (void)out_size; (void)ws_size;

    const float* in_gloss = (const float*)d_in[0];
    const float* in_clip  = (const float*)d_in[1];
    const float* in_q     = (const float*)d_in[2];
    const int*   g2c      = (const int*)d_in[3];
    const int*   g2q      = (const int*)d_in[4];
    const float* c2g_W  = (const float*)d_in[5];
    const float* c2g_a1 = (const float*)d_in[6];
    const float* c2g_a2 = (const float*)d_in[7];
    const float* g2q_W  = (const float*)d_in[8];
    const float* g2q_a1 = (const float*)d_in[9];
    const float* g2q_a2 = (const float*)d_in[10];
    const float* q2g_W  = (const float*)d_in[11];
    const float* q2g_a1 = (const float*)d_in[12];
    const float* q2g_a2 = (const float*)d_in[13];
    const float* fus_W  = (const float*)d_in[14];
    const float* fus_U  = (const float*)d_in[15];
    const float* fus_Wf = (const float*)d_in[16];
    const float* fus_Uf = (const float*)d_in[17];

    const long SG = (long)B_ * NG * FDIM;  // 2,097,152
    const long SQ = (long)B_ * NQ * FDIM;  //   524,288

    // workspace bump allocator
    char*  base = (char*)d_ws;
    size_t off  = 0;
    auto alloc = [&](size_t bytes) -> void* {
        void* p = base + off;
        off += (bytes + 255) & ~(size_t)255;
        return p;
    };
    float* cur_g = (float*)alloc(SG * 4);
    float* cur_c = (float*)alloc(SG * 4);
    float* cur_q = (float*)alloc(SQ * 4);
    float* nxt_g = (float*)alloc(SG * 4);
    float* nxt_c = (float*)alloc(SG * 4);
    float* nxt_q = (float*)alloc(SQ * 4);
    float* g1    = (float*)alloc(SG * 4);
    float* tmpA  = (float*)alloc(SG * 4);
    float* tmpF  = (float*)alloc(SG * 4);
    float* agg   = (float*)alloc(SG * 4);
    float* Wq_f  = (float*)alloc(SG * 4);
    float* Wk_f  = (float*)alloc(SG * 4);
    const long SS = (long)B_ * NHEADS_ * NG;
    float* s1v  = (float*)alloc(SS * 4);
    float* s2v  = (float*)alloc(SS * 4);
    float* cmax = (float*)alloc(SS * 4);
    float* csum = (float*)alloc(SS * 4);
    bf16* a_bf   = (bf16*)alloc(SG * 2);
    bf16* b_bf   = (bf16*)alloc(SG * 2);
    bf16* adjT   = (bf16*)alloc((long)B_ * NC * NG * 2);
    bf16* glossT = (bf16*)alloc((long)B_ * FDIM * NG * 2);
    bf16* wkt    = (bf16*)alloc((long)B_ * HD * NG * 2);
    bf16* wpack  = (bf16*)alloc((long)HD * FDIM * 2);
    bf16* wt4    = (bf16*)alloc((long)4 * FF * 2);
    bf16* attb   = (bf16*)alloc((long)B_ * NHEADS_ * NG * NC * 2);

    auto cdiv = [](long a, long b) { return (unsigned)((a + b - 1) / b); };
    auto cvt = [&](const float* in, bf16* out, long n) {
        k_f32_to_bf16<<<cdiv(n, 256), 256, 0, stream>>>(in, out, n);
    };
    auto tpose_f = [&](const float* in, bf16* out, int R, int C, long nb) {
        long tot = nb * R * C;
        k_transpose_convert<float><<<cdiv(tot, 256), 256, 0, stream>>>(in, out, R, C, tot);
    };
    auto tpose_i = [&](const int* in, bf16* out, int R, int C, long nb) {
        long tot = nb * R * C;
        k_transpose_convert<int><<<cdiv(tot, 256), 256, 0, stream>>>(in, out, R, C, tot);
    };
    auto gemm = [&](const bf16* A, const bf16* Bt, float* C, int M, int N, int K,
                    long aO, long aI, int lda, long bO, long bI, int ldb,
                    long cO, long cI, int ldc, int outer, int innerc, int acc) {
        if (N % 128 == 0) {
            dim3 grid(M / TM, N / 128, outer * innerc);
            k_gemm_bf16_nt<128><<<grid, 256, 0, stream>>>(A, Bt, C, K, aO, aI, lda,
                                                          bO, bI, ldb, cO, cI, ldc, innerc, acc);
        } else {
            dim3 grid(M / TM, N / 64, outer * innerc);
            k_gemm_bf16_nt<64><<<grid, 256, 0, stream>>>(A, Bt, C, K, aO, aI, lda,
                                                         bO, bI, ldb, cO, cI, ldc, innerc, acc);
        }
    };

    // fusion: out = sigmoid(a@Wf + b@Uf) * (a@W + b@U) + (1-f)*a
    auto fusion = [&](const float* a, const float* bsrc, int Nrows,
                      const float* W, const float* U, const float* Wf, const float* Uf,
                      float* out) {
        long ne = (long)B_ * Nrows * FDIM;
        cvt(a, a_bf, ne);
        cvt(bsrc, b_bf, ne);
        tpose_f(W,  wt4 + 0L * FF, FDIM, FDIM, 1);
        tpose_f(U,  wt4 + 1L * FF, FDIM, FDIM, 1);
        tpose_f(Wf, wt4 + 2L * FF, FDIM, FDIM, 1);
        tpose_f(Uf, wt4 + 3L * FF, FDIM, FDIM, 1);
        int Mr = B_ * Nrows;
        gemm(a_bf, wt4 + 0L * FF, tmpA, Mr, FDIM, FDIM, 0,0,FDIM, 0,0,FDIM, 0,0,FDIM, 1,1,0);
        gemm(b_bf, wt4 + 1L * FF, tmpA, Mr, FDIM, FDIM, 0,0,FDIM, 0,0,FDIM, 0,0,FDIM, 1,1,1);
        gemm(a_bf, wt4 + 2L * FF, tmpF, Mr, FDIM, FDIM, 0,0,FDIM, 0,0,FDIM, 0,0,FDIM, 1,1,0);
        gemm(b_bf, wt4 + 3L * FF, tmpF, Mr, FDIM, FDIM, 0,0,FDIM, 0,0,FDIM, 0,0,FDIM, 1,1,1);
        k_fusion_ep<<<cdiv(ne, 256), 256, 0, stream>>>(a, tmpA, tmpF, out, ne);
    };

    // GAT-style attention: q [B,Nn,F], kv [B,Mm,F], mask[n,m] via strided adj
    auto attn = [&](const float* q, int Nn, const float* kv, int Mm,
                    const int* adj, long adjB, int sN, int sM,
                    const float* W, const float* a1, const float* a2, float* outAgg) {
        k_pack_attw<<<cdiv((long)NHEADS_ * FDIM * HDIM_, 256), 256, 0, stream>>>(W, wpack);
        cvt(q,  a_bf, (long)B_ * Nn * FDIM);
        cvt(kv, b_bf, (long)B_ * Mm * FDIM);
        gemm(a_bf, wpack, Wq_f, B_ * Nn, HD, FDIM, 0,0,FDIM, 0,0,FDIM, 0,0,HD, 1,1,0);
        gemm(b_bf, wpack, Wk_f, B_ * Mm, HD, FDIM, 0,0,FDIM, 0,0,FDIM, 0,0,HD, 1,1,0);
        tpose_f(Wk_f, wkt, Mm, HD, B_);   // [B, HD, Mm]
        long tn = (long)B_ * NHEADS_ * Nn, tm = (long)B_ * NHEADS_ * Mm;
        k_dotvec<<<cdiv(tn, 256), 256, 0, stream>>>(Wq_f, a1, s1v, Nn, tn);
        k_dotvec<<<cdiv(tm, 256), 256, 0, stream>>>(Wk_f, a2, s2v, Mm, tm);
        k_colstats<<<cdiv(tm, 256), 256, 0, stream>>>(s1v, s2v, adj, adjB, sN, sM,
                                                      cmax, csum, Nn, Mm, tm);
        long ta = (long)B_ * NHEADS_ * Nn * Mm;
        k_att<<<cdiv(ta, 256), 256, 0, stream>>>(s1v, s2v, adj, adjB, sN, sM,
                                                 cmax, csum, attb, Nn, Mm, ta);
        // h[b, n, h*64+d] = sum_m att * Wk ; batched over (B, H)
        gemm(attb, wkt, outAgg, Nn, HDIM_, Mm,
             (long)NHEADS_ * Nn * Mm, (long)Nn * Mm, Mm,
             (long)HD * Mm, (long)HDIM_ * Mm, Mm,
             (long)Nn * HD, (long)HDIM_, HD, B_, NHEADS_, 0);
        long nh = (long)B_ * Nn * HD;
        k_elu<<<cdiv(nh, 256), 256, 0, stream>>>(outAgg, nh);
    };

    // ---- init ----
    hipMemcpyAsync(cur_g, in_gloss, SG * 4, hipMemcpyDeviceToDevice, stream);
    hipMemcpyAsync(cur_c, in_clip,  SG * 4, hipMemcpyDeviceToDevice, stream);
    hipMemcpyAsync(cur_q, in_q,     SQ * 4, hipMemcpyDeviceToDevice, stream);
    tpose_i(g2c, adjT, NG, NC, B_);  // adjT[b][m][n] = g2c[b][n][m]

    const long WS = (long)NHEADS_ * FDIM * HDIM_;  // per-layer attn weight slice
    const long AS = (long)NHEADS_ * HDIM_;

    for (int l = 0; l < 2; ++l) {
        // gloss_same = adjT @ gloss (per batch)
        tpose_f(cur_g, glossT, NG, FDIM, B_);
        gemm(adjT, glossT, agg, NC, FDIM, NG,
             (long)NC * NG, 0, NG, (long)FDIM * NG, 0, NG, (long)NC * FDIM, 0, FDIM, B_, 1, 0);
        // clip_1
        fusion(cur_c, agg, NC, fus_W + ((long)l * 4 + 0) * FF, fus_U + ((long)l * 4 + 0) * FF,
               fus_Wf + ((long)l * 4 + 0) * FF, fus_Uf + ((long)l * 4 + 0) * FF, nxt_c);
        // clip_agg = attn(gloss, clip, g2c)
        attn(cur_g, NG, cur_c, NC, g2c, (long)NG * NC, NC, 1,
             c2g_W + l * WS, c2g_a1 + l * AS, c2g_a2 + l * AS, agg);
        // gloss_1
        fusion(cur_g, agg, NG, fus_W + ((long)l * 4 + 1) * FF, fus_U + ((long)l * 4 + 1) * FF,
               fus_Wf + ((long)l * 4 + 1) * FF, fus_Uf + ((long)l * 4 + 1) * FF, g1);
        // gloss_agg = attn(question, gloss, g2q^T)  -> mask[n,m] = g2q[b][m][n]
        attn(cur_q, NQ, cur_g, NG, g2q, (long)NG * NQ, 1, NQ,
             g2q_W + l * WS, g2q_a1 + l * AS, g2q_a2 + l * AS, agg);
        // question_1
        fusion(cur_q, agg, NQ, fus_W + ((long)l * 4 + 2) * FF, fus_U + ((long)l * 4 + 2) * FF,
               fus_Wf + ((long)l * 4 + 2) * FF, fus_Uf + ((long)l * 4 + 2) * FF, nxt_q);
        // question_agg = attn(gloss, question, g2q)
        attn(cur_g, NG, cur_q, NQ, g2q, (long)NG * NQ, NQ, 1,
             q2g_W + l * WS, q2g_a1 + l * AS, q2g_a2 + l * AS, agg);
        // gloss_2
        fusion(g1, agg, NG, fus_W + ((long)l * 4 + 3) * FF, fus_U + ((long)l * 4 + 3) * FF,
               fus_Wf + ((long)l * 4 + 3) * FF, fus_Uf + ((long)l * 4 + 3) * FF, nxt_g);

        float* t;
        t = cur_g; cur_g = nxt_g; nxt_g = t;
        t = cur_c; cur_c = nxt_c; nxt_c = t;
        t = cur_q; cur_q = nxt_q; nxt_q = t;
    }

    float* outp = (float*)d_out;
    hipMemcpyAsync(outp,            cur_g, SG * 4, hipMemcpyDeviceToDevice, stream);
    hipMemcpyAsync(outp + SG,       cur_c, SG * 4, hipMemcpyDeviceToDevice, stream);
    hipMemcpyAsync(outp + 2 * SG,   cur_q, SQ * 4, hipMemcpyDeviceToDevice, stream);
}
